// MultiHeadAttention_39659728011763
// MI455X (gfx1250) — matrix operand order repaired and checked by reading, not compiled, not run on hardware
//
#include <hip/hip_runtime.h>
#include <math.h>

typedef __attribute__((ext_vector_type(16))) _Float16 v16h;
typedef __attribute__((ext_vector_type(8)))  _Float16 v8h;
typedef __attribute__((ext_vector_type(8)))  float    v8f;
typedef int vsi4 __attribute__((vector_size(4 * sizeof(int))));   // matches builtin param

#define D_MODEL 1024
#define DK      64
#define BT      4096      // B*T rows
#define TSEQ    2048
#define NBH     32        // B*NHEAD
#define EPSN    1e-5f
#define INV_TEMP 0.125f   // 1/sqrt(64)

#if defined(__has_builtin)
#  if __has_builtin(__builtin_amdgcn_global_load_async_to_lds_b128)
#    define HAVE_ASYNC_LDS 1
#  endif
#endif
#ifndef HAVE_ASYNC_LDS
#  define HAVE_ASYNC_LDS 0
#endif

static __device__ __forceinline__ v8f zero8() {
    v8f z;
#pragma unroll
    for (int i = 0; i < 8; ++i) z[i] = 0.0f;
    return z;
}

// A-matrix 16x32 f16 fragment: lane holds row m=lane&15.
// halves 0..7  = K in [kb + half*8,      +8)
// halves 8..15 = K in [kb + 16 + half*8, +8)
static __device__ __forceinline__ v16h make_afrag(const _Float16* row, int kb, int half) {
    v8h lo = *(const v8h*)(row + kb + half * 8);
    v8h hi = *(const v8h*)(row + kb + 16 + half * 8);
    v16h r;
#pragma unroll
    for (int i = 0; i < 8; ++i) { r[i] = lo[i]; r[i + 8] = hi[i]; }
    return r;
}

// 16-byte global->LDS copy: CDNA5 async DMA path when available.
static __device__ __forceinline__ void copy16B_to_lds(const _Float16* g, _Float16* l) {
#if HAVE_ASYNC_LDS
    __builtin_amdgcn_global_load_async_to_lds_b128(
        (__attribute__((address_space(1))) vsi4*)g,
        (__attribute__((address_space(3))) vsi4*)l, 0, 0);
#else
    *(v8h*)l = *(const v8h*)g;
#endif
}
static __device__ __forceinline__ void wait_async(int pending) {
#if HAVE_ASYNC_LDS
    if (pending) asm volatile("s_wait_asynccnt 0x2" ::: "memory");
    else         asm volatile("s_wait_asynccnt 0x0" ::: "memory");
#else
    (void)pending;
#endif
}

// ---------------- Kernel 1: ScaleNorm(x) -> f16 ----------------
__global__ void k_norm_x(const float* __restrict__ x, const float* __restrict__ scale,
                         _Float16* __restrict__ xn) {
    int row = blockIdx.x;
    const float4 v4 = ((const float4*)(x + (size_t)row * D_MODEL))[threadIdx.x];
    float s = v4.x * v4.x + v4.y * v4.y + v4.z * v4.z + v4.w * v4.w;
#pragma unroll
    for (int off = 16; off > 0; off >>= 1) s += __shfl_xor(s, off);
    __shared__ float red[8];
    if ((threadIdx.x & 31) == 0) red[threadIdx.x >> 5] = s;
    __syncthreads();
    float tot = 0.f;
#pragma unroll
    for (int i = 0; i < 8; ++i) tot += red[i];
    float f = scale[0] / fmaxf(sqrtf(tot), EPSN);
    _Float16* o = xn + (size_t)row * D_MODEL + threadIdx.x * 4;
    o[0] = (_Float16)(v4.x * f); o[1] = (_Float16)(v4.y * f);
    o[2] = (_Float16)(v4.z * f); o[3] = (_Float16)(v4.w * f);
}

// ---------------- Kernel 2: f32 -> f16 cast (weights) ----------------
__global__ void k_cast16(const float* __restrict__ w, _Float16* __restrict__ w16) {
    int i = (blockIdx.x * blockDim.x + threadIdx.x) * 4;
    const float4 v4 = *(const float4*)(w + i);
    w16[i + 0] = (_Float16)v4.x; w16[i + 1] = (_Float16)v4.y;
    w16[i + 2] = (_Float16)v4.z; w16[i + 3] = (_Float16)v4.w;
}

// ---------------- Kernel 3: C = A @ W^T + b (register double-buffered) ----------------
__global__ void k_gemm_qkv(const _Float16* __restrict__ A, const _Float16* __restrict__ W,
                           const float* __restrict__ bias, float* __restrict__ Craw) {
    int wid  = (blockIdx.x * blockDim.x + threadIdx.x) >> 5;
    int lane = threadIdx.x & 31;
    int ln = lane & 15, half = lane >> 4;
    int mb = (wid >> 4) * 16;
    int nb = (wid & 15) * 64;
    v8f acc[4] = {zero8(), zero8(), zero8(), zero8()};
    const _Float16* arow = A + (size_t)(mb + ln) * D_MODEL;
    const _Float16* wr[4];
#pragma unroll
    for (int j = 0; j < 4; ++j)
        wr[j] = W + (size_t)(nb + j * 16 + ln) * D_MODEL + half * 16;

    v16h af = *(const v16h*)(arow + half * 16);
    v16h bf[4];
#pragma unroll
    for (int j = 0; j < 4; ++j) bf[j] = *(const v16h*)(wr[j]);

    for (int kb = 0; kb < D_MODEL; kb += 32) {
        int kn = (kb + 32) & (D_MODEL - 1);        // wraps to 0 on last iter (dummy refill)
        v16h af_n = *(const v16h*)(arow + kn + half * 16);
        v16h bf_n[4];
#pragma unroll
        for (int j = 0; j < 4; ++j) bf_n[j] = *(const v16h*)(wr[j] + kn);
#pragma unroll
        for (int j = 0; j < 4; ++j)
            acc[j] = __builtin_amdgcn_wmma_f32_16x16x32_f16(
                false, af, false, bf[j], (short)0, acc[j], false, false);
        af = af_n;
#pragma unroll
        for (int j = 0; j < 4; ++j) bf[j] = bf_n[j];
    }
#pragma unroll
    for (int j = 0; j < 4; ++j) {
        float bv = bias[nb + j * 16 + ln];
#pragma unroll
        for (int r = 0; r < 8; ++r)
            Craw[(size_t)(mb + half * 8 + r) * D_MODEL + nb + j * 16 + ln] = acc[j][r] + bv;
    }
}

// ---------------- Kernel 4: row ScaleNorm -> f16 (optionally transposed for V) --------
__global__ void k_rownorm(const float* __restrict__ raw, const float* __restrict__ scale,
                          _Float16* __restrict__ out, int transposed) {
    int row = blockIdx.x;
    const float4 v4 = ((const float4*)(raw + (size_t)row * D_MODEL))[threadIdx.x];
    float s = v4.x * v4.x + v4.y * v4.y + v4.z * v4.z + v4.w * v4.w;
#pragma unroll
    for (int off = 16; off > 0; off >>= 1) s += __shfl_xor(s, off);
    __shared__ float red[8];
    if ((threadIdx.x & 31) == 0) red[threadIdx.x >> 5] = s;
    __syncthreads();
    float tot = 0.f;
#pragma unroll
    for (int i = 0; i < 8; ++i) tot += red[i];
    float f = scale[0] / fmaxf(sqrtf(tot), EPSN);
    float vals[4] = {v4.x * f, v4.y * f, v4.z * f, v4.w * f};
    int j0 = threadIdx.x * 4;
    if (!transposed) {
        _Float16* o = out + (size_t)row * D_MODEL + j0;
#pragma unroll
        for (int u = 0; u < 4; ++u) o[u] = (_Float16)vals[u];
    } else {
#pragma unroll
        for (int u = 0; u < 4; ++u) {
            unsigned fl = (unsigned)row * 1024u + (unsigned)(j0 + u);
            unsigned d = fl & 63u, tt = (fl >> 6) & 2047u, bh = fl >> 17;
            out[(size_t)bh * (DK * TSEQ) + (size_t)d * TSEQ + tt] = (_Float16)vals[u];
        }
    }
}

// ---------------- Kernel 5: row stats via S^T = K Q^T (pipelined) ----------------
__global__ void k_attn_stats(const _Float16* __restrict__ Q, const _Float16* __restrict__ K,
                             float* __restrict__ mrow, float* __restrict__ zinv) {
    int wid  = (blockIdx.x * blockDim.x + threadIdx.x) >> 5;
    int lane = threadIdx.x & 31;
    int ln = lane & 15, half = lane >> 4;
    int bh = wid >> 7;
    int tb = (wid & 127) * 16;
    const _Float16* Qb = Q + (size_t)bh * (TSEQ * DK);
    const _Float16* kbase = K + (size_t)bh * (TSEQ * DK) + (size_t)ln * DK;
    const _Float16* qrow = Qb + (size_t)(tb + ln) * DK;
    v16h q0 = make_afrag(qrow, 0, half);
    v16h q1 = make_afrag(qrow, 32, half);

    v16h a0 = make_afrag(kbase, 0, half);
    v16h a1 = make_afrag(kbase, 32, half);
    float m = -1e30f, z = 0.f;
    for (int sb = 0; sb < TSEQ; sb += 16) {
        int sn = (sb + 16) & (TSEQ - 1);           // wraps on last iter
        const _Float16* krn = kbase + (size_t)sn * DK;
        v16h n0 = make_afrag(krn, 0, half);
        v16h n1 = make_afrag(krn, 32, half);
        v8f acc = zero8();
        acc = __builtin_amdgcn_wmma_f32_16x16x32_f16(false, a0, false, q0, (short)0, acc, false, false);
        acc = __builtin_amdgcn_wmma_f32_16x16x32_f16(false, a1, false, q1, (short)0, acc, false, false);
#pragma unroll
        for (int r = 0; r < 8; ++r) {
            float v = acc[r] * INV_TEMP;
            float nm = fmaxf(m, v);
            z = z * __expf(m - nm) + __expf(v - nm);
            m = nm;
        }
        a0 = n0; a1 = n1;
    }
    float om = __shfl_xor(m, 16), oz = __shfl_xor(z, 16);
    float nm = fmaxf(m, om);
    float zz = z * __expf(m - nm) + oz * __expf(om - nm);
    if (half == 0) {
        mrow[bh * TSEQ + tb + ln] = nm;
        zinv[bh * TSEQ + tb + ln] = 1.0f / zz;
    }
}

// ---- Kernel 6: fused softmax + attn write + PV; K/V^T staged in LDS (async DMA) ------
// Block = 8 waves sharing one bh; waves take 8 consecutive t-tiles -> 8x less L2 traffic.
__global__ void k_attn_out(const _Float16* __restrict__ Q, const _Float16* __restrict__ K,
                           const _Float16* __restrict__ Vt,
                           const float* __restrict__ mrow, const float* __restrict__ zinv,
                           float* __restrict__ outO, float* __restrict__ attn) {
    __shared__ __align__(16) _Float16 lK[2][32][72];   // 32 s-rows x 64 (+8 pad)
    __shared__ __align__(16) _Float16 lV[2][64][40];   // 64 d-rows x 32 (+8 pad)

    int tid  = threadIdx.x;
    int lane = tid & 31;
    int ln = lane & 15, half = lane >> 4;
    int bh = blockIdx.x >> 4;                       // 32 bh
    int tb = ((blockIdx.x & 15) * 8 + (tid >> 5)) * 16;
    const _Float16* Qb = Q  + (size_t)bh * (TSEQ * DK);
    const _Float16* Kb = K  + (size_t)bh * (TSEQ * DK);
    const _Float16* Vb = Vt + (size_t)bh * (DK * TSEQ);
    const _Float16* qrow = Qb + (size_t)(tb + ln) * DK;
    v16h q0 = make_afrag(qrow, 0, half);
    v16h q1 = make_afrag(qrow, 32, half);
    float mt = mrow[bh * TSEQ + tb + ln];
    float zt = zinv[bh * TSEQ + tb + ln];
    v8f o[4] = {zero8(), zero8(), zero8(), zero8()};
    float* attnB = attn + ((size_t)bh * TSEQ + tb + ln) * TSEQ;

    // per-thread 16B staging slots: K tile 4KB, V tile 4KB
    int krow = tid >> 3, kc = (tid & 7) * 8;        // K: 32 rows x 128B
    int vrow = tid >> 2, vc = (tid & 3) * 8;        // V: 64 rows x 64B
    const _Float16* gK = Kb + (size_t)krow * DK + kc;
    const _Float16* gV = Vb + (size_t)vrow * TSEQ + vc;

    copy16B_to_lds(gK, &lK[0][krow][kc]);           // prologue: stage sb=0
    copy16B_to_lds(gV, &lV[0][vrow][vc]);

    int buf = 0;
    for (int sb = 0; sb < TSEQ; sb += 32) {
        int sn = sb + 32;
        if (sn < TSEQ) {                            // stage next tile into other buffer
            copy16B_to_lds(gK + (size_t)sn * DK, &lK[buf ^ 1][krow][kc]);
            copy16B_to_lds(gV + sn,              &lV[buf ^ 1][vrow][vc]);
            wait_async(1);                          // wait only for current tile's 2 asyncs
        } else {
            wait_async(0);
        }
        __syncthreads();

        v8f acc0 = zero8(), acc1 = zero8();
        {
            v16h a = make_afrag(&lK[buf][ln][0], 0, half);          // s-tile 0
            acc0 = __builtin_amdgcn_wmma_f32_16x16x32_f16(false, a, false, q0, (short)0, acc0, false, false);
            a = make_afrag(&lK[buf][ln][0], 32, half);
            acc0 = __builtin_amdgcn_wmma_f32_16x16x32_f16(false, a, false, q1, (short)0, acc0, false, false);
            a = make_afrag(&lK[buf][16 + ln][0], 0, half);          // s-tile 1
            acc1 = __builtin_amdgcn_wmma_f32_16x16x32_f16(false, a, false, q0, (short)0, acc1, false, false);
            a = make_afrag(&lK[buf][16 + ln][0], 32, half);
            acc1 = __builtin_amdgcn_wmma_f32_16x16x32_f16(false, a, false, q1, (short)0, acc1, false, false);
        }
        // S^T D-layout: lane holds fixed t, contiguous s -> directly the PV A-fragment
        v16h pa;
        float p0[8], p1[8];
#pragma unroll
        for (int r = 0; r < 8; ++r) {
            float pv0 = __expf(acc0[r] * INV_TEMP - mt) * zt;
            float pv1 = __expf(acc1[r] * INV_TEMP - mt) * zt;
            p0[r] = pv0; p1[r] = pv1;
            pa[r] = (_Float16)pv0;
            pa[r + 8] = (_Float16)pv1;
        }
        float4 s0a = {p0[0], p0[1], p0[2], p0[3]}, s0b = {p0[4], p0[5], p0[6], p0[7]};
        float4 s1a = {p1[0], p1[1], p1[2], p1[3]}, s1b = {p1[4], p1[5], p1[6], p1[7]};
        *(float4*)(attnB + sb + half * 8)          = s0a;
        *(float4*)(attnB + sb + half * 8 + 4)      = s0b;
        *(float4*)(attnB + sb + 16 + half * 8)     = s1a;
        *(float4*)(attnB + sb + 16 + half * 8 + 4) = s1b;
#pragma unroll
        for (int j = 0; j < 4; ++j) {
            v16h bf = make_afrag(&lV[buf][j * 16 + ln][0], 0, half);
            o[j] = __builtin_amdgcn_wmma_f32_16x16x32_f16(false, pa, false, bf, (short)0, o[j], false, false);
        }
        __syncthreads();                            // tile fully consumed before re-stage
        buf ^= 1;
    }
#pragma unroll
    for (int j = 0; j < 4; ++j) {
        float* ob = outO + (size_t)bh * (DK * TSEQ) + (size_t)(j * 16 + ln) * TSEQ + tb + half * 8;
        float4 lo4 = {o[j][0], o[j][1], o[j][2], o[j][3]};
        float4 hi4 = {o[j][4], o[j][5], o[j][6], o[j][7]};
        *(float4*)ob = lo4;
        *(float4*)(ob + 4) = hi4;
    }
}

extern "C" void kernel_launch(void* const* d_in, const int* in_sizes, int n_in,
                              void* d_out, int out_size, void* d_ws, size_t ws_size,
                              hipStream_t stream) {
    const float* x     = (const float*)d_in[0];
    const float* scale = (const float*)d_in[1];
    const float* w_qs  = (const float*)d_in[2];
    const float* b_qs  = (const float*)d_in[3];
    const float* w_ks  = (const float*)d_in[4];
    const float* b_ks  = (const float*)d_in[5];
    const float* w_vs  = (const float*)d_in[6];
    const float* b_vs  = (const float*)d_in[7];
    float* out  = (float*)d_out;
    float* attn = out + (size_t)2 * TSEQ * D_MODEL;

    char* ws = (char*)d_ws;
    size_t off = 0;
    auto alloc = [&](size_t bytes) {
        void* p = ws + off;
        off += (bytes + 255) & ~(size_t)255;
        return p;
    };
    _Float16* xn16 = (_Float16*)alloc((size_t)BT * D_MODEL * 2);
    _Float16* wq16 = (_Float16*)alloc((size_t)D_MODEL * D_MODEL * 2);
    _Float16* wk16 = (_Float16*)alloc((size_t)D_MODEL * D_MODEL * 2);
    _Float16* wv16 = (_Float16*)alloc((size_t)D_MODEL * D_MODEL * 2);
    float*    qraw = (float*)   alloc((size_t)BT * D_MODEL * 4);
    float*    kraw = (float*)   alloc((size_t)BT * D_MODEL * 4);
    float*    vraw = (float*)   alloc((size_t)BT * D_MODEL * 4);
    _Float16* q16  = (_Float16*)alloc((size_t)BT * D_MODEL * 2);
    _Float16* k16  = (_Float16*)alloc((size_t)BT * D_MODEL * 2);
    _Float16* vt16 = (_Float16*)alloc((size_t)BT * D_MODEL * 2);
    float*    mrow = (float*)   alloc((size_t)NBH * TSEQ * 4);
    float*    zinv = (float*)   alloc((size_t)NBH * TSEQ * 4);

    dim3 blk(256);
    k_norm_x<<<BT, blk, 0, stream>>>(x, scale, xn16);
    k_cast16<<<D_MODEL * D_MODEL / 1024, blk, 0, stream>>>(w_qs, wq16);
    k_cast16<<<D_MODEL * D_MODEL / 1024, blk, 0, stream>>>(w_ks, wk16);
    k_cast16<<<D_MODEL * D_MODEL / 1024, blk, 0, stream>>>(w_vs, wv16);
    k_gemm_qkv<<<512, blk, 0, stream>>>(xn16, wq16, b_qs, qraw);
    k_gemm_qkv<<<512, blk, 0, stream>>>(xn16, wk16, b_ks, kraw);
    k_gemm_qkv<<<512, blk, 0, stream>>>(xn16, wv16, b_vs, vraw);
    k_rownorm<<<BT, blk, 0, stream>>>(qraw, scale, q16, 0);
    k_rownorm<<<BT, blk, 0, stream>>>(kraw, scale, k16, 0);
    k_rownorm<<<BT, blk, 0, stream>>>(vraw, scale, vt16, 1);
    k_attn_stats<<<512, blk, 0, stream>>>(q16, k16, mrow, zinv);
    k_attn_out<<<512, blk, 0, stream>>>(q16, k16, vt16, mrow, zinv, out, attn);
}
